// MultiHeadsDotAttention_15135464751442
// MI455X (gfx1250) — compile-verified, matrix operand order checked by reading, not run on hardware
//
#include <hip/hip_runtime.h>

typedef unsigned short u16;
typedef unsigned int   u32;
typedef unsigned long long u64;
typedef __attribute__((ext_vector_type(16))) __bf16 v16bf;
typedef __attribute__((ext_vector_type(8)))  float  v8f;
typedef __attribute__((ext_vector_type(4)))  float  v4f;
typedef __attribute__((ext_vector_type(8)))  u16    us8;
typedef __attribute__((ext_vector_type(4)))  u16    us4;
typedef __attribute__((ext_vector_type(4)))  u32    v4u;
typedef __attribute__((ext_vector_type(8)))  u32    v8u;

#define D_MODEL 1024
#define N_HEADS 16
#define D_K     64
#define BATCH   4
#define SEQ     1024
#define M_TOT   (BATCH * SEQ)

__device__ __forceinline__ u16 f2bf(float f) {
  u32 u = __builtin_bit_cast(u32, f);
  u32 r = u + 0x7FFFu + ((u >> 16) & 1u);   // round-to-nearest-even
  return (u16)(r >> 16);
}

union FragU { v16bf v; us8 h[2]; };

// 16-element bf16 fragment for the CDNA5 16-bit A/B layout: lane holds row
// (M%16 / N%16); k-chunks [off, off+8) and [off+16, off+24), off = half*8
// (+ kstep*32).  Two 16-byte LDS loads.
__device__ __forceinline__ v16bf load_frag(const u16* rowPtr, int off) {
  FragU f;
  f.h[0] = *(const us8*)(rowPtr + off);
  f.h[1] = *(const us8*)(rowPtr + off + 16);
  return f.v;
}

// Raw LDS byte offset of a __shared__ object (async-to-LDS VDST / TDM lds_addr).
typedef __attribute__((address_space(3))) const char* lds_cptr_t;
__device__ __forceinline__ u32 lds_off(const void* p) {
  return (u32)(size_t)(lds_cptr_t)p;
}

// GLOBAL_LOAD_ASYNC_TO_LDS_B128, GVS mode: mem = s[base] + v_off, dest LDS+l.
__device__ __forceinline__ void async_copy_b128(void* ldsDst, const void* gbase,
                                                u32 voffBytes) {
  u32 l = lds_off(ldsDst);
  asm volatile("global_load_async_to_lds_b128 %0, %1, %2"
               :: "v"(l), "v"(voffBytes), "s"((u64)(size_t)gbase)
               : "memory");
}
__device__ __forceinline__ void wait_async_all() {
  asm volatile("s_wait_asynccnt 0x0" ::: "memory");
}

// ---------------------------------------------------------------------------
// Tensor Data Mover: load a 64-row x 128B bf16 tile (row stride 128B in
// memory) into LDS with a 16B pad after every 128B row (-> 144B LDS stride).
// D# layout per CDNA5 ISA ch.8: group0 = {flags, lds_addr, global_addr, type},
// group1 = {pad/data_size, tensor dims, tile dims, strides}.  Issued by one
// wave; tracked by TENSORcnt.
// ---------------------------------------------------------------------------
__device__ __forceinline__ void tdm_load_k_tile(u32 ldsBase, const void* gsrc) {
  u64 ga = (u64)(size_t)gsrc;
  v4u g0 = { 1u,                                   // count=1 (valid user D#)
             ldsBase,                              // lds_addr [63:32]
             (u32)ga,                              // global_addr[31:0]
             ((u32)(ga >> 32) & 0x01FFFFFFu) | (2u << 30) };  // ga[56:32]|type=2
  const u32 w0 = (1u << 16)      // data_size = 2 bytes
               | (1u << 20)      // pad_enable
               | (4u << 22)      // pad_interval: 32 DWORDs (=128B) between pads
               | (3u << 25);     // pad_amount: 4 DWORDs (=16B)
  v8u g1 = { w0,
             (u32)(64u << 16),   // tensor_dim0 = 64 elems  (bits [79:48])
             (u32)(1024u << 16), // tensor_dim1 = 1024 rows (bits [111:80])
             (u32)(64u << 16),   // tile_dim0 = 64          (bits [127:112])
             64u,                // tile_dim1 = 64, tile_dim2 = 0
             64u,                // tensor_dim0_stride = 64 elems
             0u, 0u };           // tensor_dim1_stride = 0 (unused, 2D tile)
  asm volatile("tensor_load_to_lds %0, %1" :: "s"(g0), "s"(g1) : "memory");
}

__device__ __forceinline__ float rowmax16(float v) {
#pragma unroll
  for (int m = 1; m < 16; m <<= 1) v = fmaxf(v, __shfl_xor(v, m, 32));
  return v;
}
__device__ __forceinline__ float rowsum16(float v) {
#pragma unroll
  for (int m = 1; m < 16; m <<= 1) v += __shfl_xor(v, m, 32);
  return v;
}

// ---------------------------------------------------------------------------
// Fill the attn_score region with -inf (upper-triangular tiles keep it).
// ---------------------------------------------------------------------------
__global__ __launch_bounds__(256) void fill_neginf(v4f* __restrict__ p, int n4) {
  int idx = blockIdx.x * blockDim.x + threadIdx.x;
  const float ninf = -__builtin_inff();
  v4f v = { ninf, ninf, ninf, ninf };
  for (int i = idx; i < n4; i += (int)(gridDim.x * blockDim.x)) p[i] = v;
}

// ---------------------------------------------------------------------------
// Y = X @ W^T + b   (M=4096, N=1024, K=1024), bf16 WMMA with f32 accumulate.
// IN_BF16: X is bf16 (workspace, staged via async-to-LDS) vs f32 (converted).
// OUT_MODE 0: bf16 to head-split [B][H][P][64] workspace (Q/K/V)
// OUT_MODE 1: f32 to plain [M][N] (final output)
// Compile-time flags keep the K-loop free of SALU flag checks.
// ---------------------------------------------------------------------------
template <bool IN_BF16, int OUT_MODE>
__global__ __launch_bounds__(256) void proj_gemm(
    const void* __restrict__ Xin, const float* __restrict__ W,
    const float* __restrict__ bias, void* __restrict__ Y)
{
  __shared__ __align__(16) u16 At[128][40];   // +8 pad, 80B row stride
  __shared__ __align__(16) u16 Bt[64][40];

  const int tid  = threadIdx.x;
  const int wave = tid >> 5, lane = tid & 31;
  const int m16  = lane & 15, half = lane >> 4;
  const int mBase = blockIdx.x * 128;
  const int nBase = blockIdx.y * 64;

  v8f acc[4] = {};

  const float* Xf = (const float*)Xin;
  const u16*   Xb = (const u16*)Xin;

  for (int kb = 0; kb < D_MODEL; kb += 32) {
    // --- stage A tile: 128 x 32 ---
    if constexpr (IN_BF16) {
      int r0 = tid >> 2, c = (tid & 3) * 8;
      const u16* base = &Xb[(size_t)mBase * D_MODEL + kb];
#pragma unroll
      for (int it = 0; it < 2; ++it) {
        int r = r0 + it * 64;
        async_copy_b128(&At[r][c], base, (u32)((r * D_MODEL + c) * 2));
      }
    } else {
      int r0 = tid >> 3, c = (tid & 7) * 4;
#pragma unroll
      for (int it = 0; it < 4; ++it) {
        int r = r0 + it * 32;
        const float* gp = &Xf[(size_t)(mBase + r) * D_MODEL + kb + c];
        if (kb + 32 < D_MODEL) __builtin_prefetch(gp + 32, 0, 3);
        v4f x = *(const v4f*)gp;
        us4 o = { f2bf(x[0]), f2bf(x[1]), f2bf(x[2]), f2bf(x[3]) };
        *(us4*)&At[r][c] = o;
      }
    }
    // --- stage B tile: 64 x 32 (weight rows, k-contiguous like A) ---
    {
      int r0 = tid >> 3, c = (tid & 7) * 4;
#pragma unroll
      for (int it = 0; it < 2; ++it) {
        int r = r0 + it * 32;
        const float* gp = &W[(size_t)(nBase + r) * D_MODEL + kb + c];
        if (kb + 32 < D_MODEL) __builtin_prefetch(gp + 32, 0, 3);
        v4f x = *(const v4f*)gp;
        us4 o = { f2bf(x[0]), f2bf(x[1]), f2bf(x[2]), f2bf(x[3]) };
        *(us4*)&Bt[r][c] = o;
      }
    }
    if constexpr (IN_BF16) wait_async_all();
    __syncthreads();

    v16bf fa = load_frag(&At[wave * 16 + m16][0], half * 8);
#pragma unroll
    for (int t = 0; t < 4; ++t) {
      v16bf fb = load_frag(&Bt[t * 16 + m16][0], half * 8);
      acc[t] = __builtin_amdgcn_wmma_f32_16x16x32_bf16(
          false, fa, false, fb, (short)0, acc[t], false, false);
    }
    __syncthreads();
  }

  // epilogue: bias + store  (C layout: lane l -> n = l%16, m = r + 8*(l/16))
#pragma unroll
  for (int t = 0; t < 4; ++t) {
    int n = nBase + t * 16 + m16;
    float bv = bias[n];
#pragma unroll
    for (int r = 0; r < 8; ++r) {
      int m = mBase + wave * 16 + r + 8 * half;
      float val = acc[t][r] + bv;
      if constexpr (OUT_MODE == 0) {
        int bb = m >> 10, p = m & 1023, hh = n >> 6, d = n & 63;
        ((u16*)Y)[((size_t)((bb * N_HEADS + hh) * SEQ + p)) * D_K + d] = f2bf(val);
      } else {
        ((float*)Y)[(size_t)m * D_MODEL + n] = val;
      }
    }
  }
}

// ---------------------------------------------------------------------------
// Causal attention per (b, h, 128-row strip): S = QK^T/8 (raw masked scores
// to d_out), online softmax, O = softmax(S) @ V.  Flash-style j loop.
// Q streams via GLOBAL_LOAD_ASYNC_TO_LDS_B128 (ASYNCcnt); K tiles stream via
// the Tensor Data Mover (TENSORcnt, one wave issues the DMA); the V transpose
// (VGPR work) overlaps with the in-flight DMA.
// ---------------------------------------------------------------------------
__global__ __launch_bounds__(256) void attn_kernel(
    const u16* __restrict__ Qw, const u16* __restrict__ Kw,
    const u16* __restrict__ Vw, float* __restrict__ scores,
    u16* __restrict__ Ow)
{
  __shared__ __align__(16) u16 Qs[128][72];   // 128 i-rows x 64 d
  __shared__ __align__(16) u16 Ks[64][72];    // 64 j-rows  x 64 d (TDM pads rows)
  __shared__ __align__(16) u16 Vt[64][72];    // transposed: 64 d-rows x 64 j
  __shared__ __align__(16) u16 Ps[128][72];   // probs, per-wave 16-row strips

  const int tid  = threadIdx.x;
  const int wave = tid >> 5, lane = tid & 31;
  const int m16  = lane & 15, half = lane >> 4;
  const int i0 = blockIdx.x * 128;
  const int h  = blockIdx.y;
  const int b  = blockIdx.z;
  const size_t bhBase = (size_t)(b * N_HEADS + h) * SEQ;

  // load Q strip once (async copy, no VGPR round-trip)
  {
    int r0 = tid >> 3, c = (tid & 7) * 8;
    const u16* base = &Qw[(bhBase + i0) * D_K];
#pragma unroll
    for (int it = 0; it < 4; ++it) {
      int r = r0 + it * 32;
      async_copy_b128(&Qs[r][c], base, (u32)((r * D_K + c) * 2));
    }
  }
  wait_async_all();
  __syncthreads();

  v16bf fq[2];
  fq[0] = load_frag(&Qs[wave * 16 + m16][0], half * 8);
  fq[1] = load_frag(&Qs[wave * 16 + m16][0], 32 + half * 8);

  v8f o[4] = {};
  float run_m[8], run_l[8];
#pragma unroll
  for (int r = 0; r < 8; ++r) { run_m[r] = -__builtin_inff(); run_l[r] = 0.f; }

  const int jEnd = i0 + 128;                 // uniform per block -> legal barriers
  for (int jb = 0; jb < jEnd; jb += 64) {
    __syncthreads();                          // protect prior Ks/Vt readers
    // stage K tile via Tensor Data Mover (one DMA for the whole tile)
    if (wave == 0) {
      tdm_load_k_tile(lds_off(&Ks[0][0]), &Kw[(bhBase + jb) * D_K]);
    }
    // stage V tile transposed in LDS (VGPR work; overlaps the TDM)
    {
      int r0 = tid >> 3, c = (tid & 7) * 8;
#pragma unroll
      for (int it = 0; it < 2; ++it) {
        int j = r0 + it * 32;
        us8 v = *(const us8*)&Vw[(bhBase + jb + j) * D_K + c];
#pragma unroll
        for (int q = 0; q < 8; ++q) Vt[c + q][j] = v[q];
      }
    }
    if (wave == 0) __builtin_amdgcn_s_wait_tensorcnt(0);
    __syncthreads();

    // S tile: this wave's 16 rows x 64 cols, k = d (2 WMMA k-steps)
    v8f s[4] = {};
#pragma unroll
    for (int jt = 0; jt < 4; ++jt) {
#pragma unroll
      for (int kk = 0; kk < 2; ++kk) {
        v16bf fk = load_frag(&Ks[jt * 16 + m16][0], kk * 32 + half * 8);
        s[jt] = __builtin_amdgcn_wmma_f32_16x16x32_bf16(
            false, fq[kk], false, fk, (short)0, s[jt], false, false);
      }
    }

    // scale, causal mask, emit raw scores
    const int iRow = i0 + wave * 16;
#pragma unroll
    for (int jt = 0; jt < 4; ++jt) {
      int j_g = jb + jt * 16 + m16;
#pragma unroll
      for (int r = 0; r < 8; ++r) {
        int i_g = iRow + r + 8 * half;
        float val = s[jt][r] * 0.125f;        // 1/sqrt(64)
        if (j_g > i_g) val = -__builtin_inff();
        s[jt][r] = val;
        scores[(bhBase + i_g) * SEQ + j_g] = val;
      }
    }

    // online softmax (row reductions across the 16 n-lanes of each half)
#pragma unroll
    for (int r = 0; r < 8; ++r) {
      float tm = fmaxf(fmaxf(s[0][r], s[1][r]), fmaxf(s[2][r], s[3][r]));
      tm = rowmax16(tm);
      float nm = fmaxf(run_m[r], tm);
      float alpha = __expf(run_m[r] - nm);    // -inf history -> 0; equal -> 1
      run_m[r] = nm;
      float psum = 0.f;
      int prow = wave * 16 + r + 8 * half;
#pragma unroll
      for (int jt = 0; jt < 4; ++jt) {
        float p = __expf(s[jt][r] - nm);      // masked -inf -> 0
        psum += p;
        Ps[prow][jt * 16 + m16] = f2bf(p);
      }
      psum = rowsum16(psum);
      run_l[r] = run_l[r] * alpha + psum;
#pragma unroll
      for (int dt = 0; dt < 4; ++dt) o[dt][r] *= alpha;   // rescale O accum
    }

    // O += P @ V   (A = probs 16xK from Ps, B = Vt columns; K = 64)
#pragma unroll
    for (int kk = 0; kk < 2; ++kk) {
      v16bf fp = load_frag(&Ps[wave * 16 + m16][0], kk * 32 + half * 8);
#pragma unroll
      for (int dt = 0; dt < 4; ++dt) {
        v16bf fv = load_frag(&Vt[dt * 16 + m16][0], kk * 32 + half * 8);
        o[dt] = __builtin_amdgcn_wmma_f32_16x16x32_bf16(
            false, fp, false, fv, (short)0, o[dt], false, false);
      }
    }
  }

  // normalize and store O (merged-head [B][P][D_MODEL] bf16)
#pragma unroll
  for (int r = 0; r < 8; ++r) {
    float inv = 1.f / run_l[r];
    int i_g = i0 + wave * 16 + r + 8 * half;
#pragma unroll
    for (int dt = 0; dt < 4; ++dt) {
      float val = o[dt][r] * inv;
      Ow[((size_t)b * SEQ + i_g) * D_MODEL + h * D_K + dt * 16 + m16] = f2bf(val);
    }
  }
}

// ---------------------------------------------------------------------------
extern "C" void kernel_launch(void* const* d_in, const int* in_sizes, int n_in,
                              void* d_out, int out_size, void* d_ws, size_t ws_size,
                              hipStream_t stream) {
  (void)in_sizes; (void)n_in; (void)out_size; (void)ws_size;
  const float* query = (const float*)d_in[0];
  const float* key_  = (const float*)d_in[1];
  const float* value = (const float*)d_in[2];
  const float* Wq = (const float*)d_in[3];
  const float* bq = (const float*)d_in[4];
  const float* Wk = (const float*)d_in[5];
  const float* bk = (const float*)d_in[6];
  const float* Wv = (const float*)d_in[7];
  const float* bv = (const float*)d_in[8];
  const float* Wo = (const float*)d_in[9];
  const float* bo = (const float*)d_in[10];

  float* out    = (float*)d_out;
  float* scores = out + (size_t)M_TOT * D_MODEL;          // after 4M f32 of `out`

  u16* wsq = (u16*)d_ws;                                  // bf16 Q [B][H][P][64]
  u16* wsk = wsq + (size_t)M_TOT * D_MODEL;               // bf16 K
  u16* wsv = wsk + (size_t)M_TOT * D_MODEL;               // bf16 V
  u16* wso = wsv + (size_t)M_TOT * D_MODEL;               // bf16 O [B][P][1024]

  const int score_v4 = (int)((size_t)BATCH * N_HEADS * SEQ * SEQ / 4);
  fill_neginf<<<8192, 256, 0, stream>>>((v4f*)scores, score_v4);

  dim3 g(32, 16, 1);
  proj_gemm<false, 0><<<g, 256, 0, stream>>>(query, Wq, bq, wsq);
  proj_gemm<false, 0><<<g, 256, 0, stream>>>(key_,  Wk, bk, wsk);
  proj_gemm<false, 0><<<g, 256, 0, stream>>>(value, Wv, bv, wsv);

  attn_kernel<<<dim3(8, 16, 4), 256, 0, stream>>>(wsq, wsk, wsv, scores, wso);

  proj_gemm<true, 1><<<g, 256, 0, stream>>>(wso, Wo, bo, out);
}
